// RWKV_Tmix_x060_54906861912136
// MI455X (gfx1250) — matrix operand order repaired and checked by reading, not compiled, not run on hardware
//
#include <hip/hip_runtime.h>
#include <hip/hip_bf16.h>

// ---------------------------------------------------------------------------
// RWKV6 time-mix (x060) for MI455X / gfx1250, wave32 + WMMA bf16.
// Shapes: B=4, T=2048, C=2048, H=32, N=64, D_MIX=32 (x5), D_DECAY=64.
// ---------------------------------------------------------------------------

#define RW_B     4
#define RW_T     2048
#define RW_C     2048
#define RW_H     32
#define RW_N     64
#define RW_BT    (RW_B * RW_T)          // 8192
#define RW_DMIX5 160                    // 5 * 32
#define RW_DDEC  64
#define RW_EPS   6.4e-4f                // 1e-5 * 8^2

typedef __attribute__((ext_vector_type(16))) __bf16 v16bf;
typedef __attribute__((ext_vector_type(8)))  float  v8f;
typedef __attribute__((__vector_size__(4 * sizeof(int)))) int v4i_t;

// CDNA5 async global->LDS copy (ASYNCcnt) if this toolchain exposes it.
#if __has_builtin(__builtin_amdgcn_global_load_async_to_lds_b128) && \
    __has_builtin(__builtin_amdgcn_s_wait_asynccnt)
#define USE_ASYNC_LDS 1
#else
#define USE_ASYNC_LDS 0
#endif

__device__ __forceinline__ __bf16 f2bf(float f) {
  union { float f; unsigned u; } a; a.f = f;
  unsigned r = a.u + 0x7FFFu + ((a.u >> 16) & 1u);       // round-to-nearest-even
  union { unsigned short s; __bf16 b; } o; o.s = (unsigned short)(r >> 16);
  return o.b;
}

union FragBF { uint4 q[2]; v16bf v; };

// ---------------------------------------------------------------------------
// WMMA bf16 GEMM:  C[M,N] = epi( A[M,K] * B[N,K]^T + bias[N] )
// Block tile 128x256 (grid.y x grid.x), 8 waves of 32 -> 64x64 per-wave tile.
// K staged 32 at a time through double-buffered LDS.
// EPI: 0 = none, 1 = tanh, 2 = silu.  EXACT: M%128==0 && N%256==0 (no guards).
// ---------------------------------------------------------------------------
template <int EPI, bool EXACT>
__global__ __launch_bounds__(256) void gemm_bf16_kernel(
    const __bf16* __restrict__ A, const __bf16* __restrict__ Bw,
    float* __restrict__ C, const float* __restrict__ bias,
    int M, int N, int K) {
  __shared__ __bf16 sA[2][128][40];   // 32 K + 8 pad  (20 KB)
  __shared__ __bf16 sB[2][256][40];   //               (40 KB)

  const int tid  = threadIdx.x;
  const int lane = tid & 31;
  const int wave = tid >> 5;
  const int wm   = wave >> 2;   // 0..1
  const int wn   = wave & 3;    // 0..3
  const int m0   = blockIdx.y * 128;
  const int n0   = blockIdx.x * 256;

  v8f acc[4][4];
#pragma unroll
  for (int i = 0; i < 4; ++i)
#pragma unroll
    for (int j = 0; j < 4; ++j) acc[i][j] = (v8f)(0.0f);

  auto stage = [&](int kk, int buf) {
    if (EXACT) {
#if USE_ASYNC_LDS
      // Async DMA copy: no VGPR data movement, tracked by ASYNCcnt.
#pragma unroll
      for (int it = 0; it < 2; ++it) {
        int idx = tid + it * 256;
        int row = idx >> 2, ch = idx & 3;
        __builtin_amdgcn_global_load_async_to_lds_b128(
            (v4i_t*)(void*)(A + (size_t)(m0 + row) * K + kk + ch * 8),
            (v4i_t*)(void*)&sA[buf][row][ch * 8],
            0, 0);
      }
#pragma unroll
      for (int it = 0; it < 4; ++it) {
        int idx = tid + it * 256;
        int row = idx >> 2, ch = idx & 3;
        __builtin_amdgcn_global_load_async_to_lds_b128(
            (v4i_t*)(void*)(Bw + (size_t)(n0 + row) * K + kk + ch * 8),
            (v4i_t*)(void*)&sB[buf][row][ch * 8],
            0, 0);
      }
#else
#pragma unroll
      for (int it = 0; it < 2; ++it) {
        int idx = tid + it * 256;
        int row = idx >> 2, ch = idx & 3;
        uint4 d = *(const uint4*)(A + (size_t)(m0 + row) * K + kk + ch * 8);
        *(uint4*)&sA[buf][row][ch * 8] = d;
      }
#pragma unroll
      for (int it = 0; it < 4; ++it) {
        int idx = tid + it * 256;
        int row = idx >> 2, ch = idx & 3;
        uint4 d = *(const uint4*)(Bw + (size_t)(n0 + row) * K + kk + ch * 8);
        *(uint4*)&sB[buf][row][ch * 8] = d;
      }
#endif
    } else {
      // Guarded path (partial N tiles): zero-fill out-of-range rows.
#pragma unroll
      for (int it = 0; it < 2; ++it) {
        int idx = tid + it * 256;
        int row = idx >> 2, ch = idx & 3;
        uint4 d = make_uint4(0u, 0u, 0u, 0u);
        int gr = m0 + row;
        if (gr < M) d = *(const uint4*)(A + (size_t)gr * K + kk + ch * 8);
        *(uint4*)&sA[buf][row][ch * 8] = d;
      }
#pragma unroll
      for (int it = 0; it < 4; ++it) {
        int idx = tid + it * 256;
        int row = idx >> 2, ch = idx & 3;
        uint4 d = make_uint4(0u, 0u, 0u, 0u);
        int gr = n0 + row;
        if (gr < N) d = *(const uint4*)(Bw + (size_t)gr * K + kk + ch * 8);
        *(uint4*)&sB[buf][row][ch * 8] = d;
      }
    }
  };

  auto fence_stage = [&]() {
#if USE_ASYNC_LDS
    if (EXACT) __builtin_amdgcn_s_wait_asynccnt(0);
#endif
  };

  stage(0, 0);
  fence_stage();
  __syncthreads();

  int buf = 0;
  for (int kk = 0; kk < K; kk += 32) {
    if (kk + 32 < K) stage(kk + 32, buf ^ 1);
    if (EXACT && kk + 64 < K) {              // L2 prefetch two stages ahead
      __builtin_prefetch(A + (size_t)(m0 + (tid >> 1)) * K + kk + 64, 0, 0);
      __builtin_prefetch(Bw + (size_t)(n0 + tid) * K + kk + 64, 0, 0);
    }

    // Fragment loads per CDNA5 16-bit A/B VGPR layouts (ISA 7.12.2)
    FragBF af[4], bfm[4];
    const int rl  = lane & 15;
    const int akb = (lane < 16) ? 0 : 8;    // A: lanes 16-31 hold K 8..15 / 24..31
    const int bkb = (lane < 16) ? 0 : 8;    // B: lanes 16-31 hold K 8..15 / 24..31
#pragma unroll
    for (int mt = 0; mt < 4; ++mt) {
      int row = wm * 64 + mt * 16 + rl;
      af[mt].q[0] = *(const uint4*)&sA[buf][row][akb];
      af[mt].q[1] = *(const uint4*)&sA[buf][row][akb + 16];
    }
#pragma unroll
    for (int nt = 0; nt < 4; ++nt) {
      int row = wn * 64 + nt * 16 + rl;
      bfm[nt].q[0] = *(const uint4*)&sB[buf][row][bkb];
      bfm[nt].q[1] = *(const uint4*)&sB[buf][row][bkb + 16];
    }
#pragma unroll
    for (int mt = 0; mt < 4; ++mt)
#pragma unroll
      for (int nt = 0; nt < 4; ++nt)
        acc[mt][nt] = __builtin_amdgcn_wmma_f32_16x16x32_bf16(
            false, af[mt].v, false, bfm[nt].v, (short)0, acc[mt][nt],
            false, false);

    fence_stage();
    __syncthreads();
    buf ^= 1;
  }

  // Epilogue: C/D layout -> lane (l&15) = column, VGPR i = row (+8 for l>=16)
#pragma unroll
  for (int mt = 0; mt < 4; ++mt) {
#pragma unroll
    for (int nt = 0; nt < 4; ++nt) {
      int col = n0 + wn * 64 + nt * 16 + (lane & 15);
      if (EXACT || col < N) {
        float bv = bias ? bias[col] : 0.0f;
        int rbase = m0 + wm * 64 + mt * 16 + ((lane < 16) ? 0 : 8);
#pragma unroll
        for (int i = 0; i < 8; ++i) {
          float v = acc[mt][nt][i] + bv;
          if (EPI == 1) v = tanhf(v);
          if (EPI == 2) v = v / (1.0f + __expf(-v));
          C[(size_t)(rbase + i) * N + col] = v;
        }
      }
    }
  }
}

// ---------------------------------------------------------------------------
// Elementwise / conversion kernels
// ---------------------------------------------------------------------------
__global__ void cvt_f32_to_bf16_kernel(const float* __restrict__ in,
                                       __bf16* __restrict__ out, long n) {
  long i = (long)blockIdx.x * blockDim.x + threadIdx.x;
  if (i < n) out[i] = f2bf(in[i]);
}

// in[rows][cols] -> out[cols][rows]  (makes K contiguous for the GEMM B operand)
__global__ void cvtT_f32_to_bf16_kernel(const float* __restrict__ in,
                                        __bf16* __restrict__ out,
                                        int rows, int cols) {
  long i = (long)blockIdx.x * blockDim.x + threadIdx.x;
  long n = (long)rows * cols;
  if (i < n) {
    int r = (int)(i / cols), c = (int)(i % cols);
    out[(size_t)c * rows + r] = f2bf(in[i]);
  }
}

// xxx = x + (shift(x) - x) * time_maa_x, in bf16 for the mix GEMM
__global__ void prep_xxx_kernel(const float* __restrict__ x,
                                const float* __restrict__ maa_x,
                                __bf16* __restrict__ xxx, long total) {
  long i = (long)blockIdx.x * blockDim.x + threadIdx.x;
  if (i >= total) return;
  int  c  = (int)(i % RW_C);
  long bt = i / RW_C;
  int  t  = (int)(bt % RW_T);
  float xc   = x[i];
  float prev = (t > 0) ? x[i - RW_C] : 0.0f;
  xxx[i] = f2bf(xc + (prev - xc) * maa_x[c]);
}

// m[f] = a_row . w2[f], then the 5 lerped inputs in bf16
__global__ __launch_bounds__(256) void mix5_kernel(
    const float* __restrict__ x, const float* __restrict__ a,
    const float* __restrict__ w2,
    const float* __restrict__ maa_w, const float* __restrict__ maa_k,
    const float* __restrict__ maa_v, const float* __restrict__ maa_r,
    const float* __restrict__ maa_g,
    __bf16* __restrict__ xw, __bf16* __restrict__ xk, __bf16* __restrict__ xv,
    __bf16* __restrict__ xr, __bf16* __restrict__ xg) {
  int bt = blockIdx.x;
  int t  = bt % RW_T;
  __shared__ float sa[RW_DMIX5];
  if (threadIdx.x < RW_DMIX5) sa[threadIdx.x] = a[(size_t)bt * RW_DMIX5 + threadIdx.x];
  __syncthreads();
  for (int c = threadIdx.x; c < RW_C; c += 256) {
    size_t idx = (size_t)bt * RW_C + c;
    float xc   = x[idx];
    float prev = (t > 0) ? x[idx - RW_C] : 0.0f;
    float xx   = prev - xc;
    float m[5];
#pragma unroll
    for (int f = 0; f < 5; ++f) {
      float s = 0.0f;
#pragma unroll
      for (int d = 0; d < 32; ++d)
        s += sa[f * 32 + d] * w2[(size_t)(f * 32 + d) * RW_C + c];
      m[f] = s;
    }
    xw[idx] = f2bf(xc + xx * (maa_w[c] + m[0]));
    xk[idx] = f2bf(xc + xx * (maa_k[c] + m[1]));
    xv[idx] = f2bf(xc + xx * (maa_v[c] + m[2]));
    xr[idx] = f2bf(xc + xx * (maa_r[c] + m[3]));
    xg[idx] = f2bf(xc + xx * (maa_g[c] + m[4]));
  }
}

// ---------------------------------------------------------------------------
// WKV6 sequential scan. One block per (b,h); 64x64 f32 state in registers
// (16 per thread), per-step r/k/v/decay vectors broadcast via LDS.
// ---------------------------------------------------------------------------
__global__ __launch_bounds__(256) void wkv6_scan_kernel(
    const float* __restrict__ r, const float* __restrict__ k,
    const float* __restrict__ v, const float* __restrict__ w,
    const float* __restrict__ u, float* __restrict__ y) {
  const int bh = blockIdx.x;
  const int b  = bh / RW_H, h = bh % RW_H;
  const int tid = threadIdx.x;
  const int i   = tid & 63;     // state column / output index
  const int jg  = tid >> 6;     // j group (0..3), 16 j's per thread

  __shared__ float rt[64], kt[64], vt[64], dt[64];
  __shared__ float red[4][64];

  float S[16], uj[16];
#pragma unroll
  for (int jj = 0; jj < 16; ++jj) {
    S[jj]  = 0.0f;
    uj[jj] = u[h * 64 + jg * 16 + jj];
  }

  const size_t base = (size_t)b * RW_T * RW_C + (size_t)h * 64;
  for (int t = 0; t < RW_T; ++t) {
    size_t off = base + (size_t)t * RW_C;
    if (tid < 64)       rt[tid]       = r[off + tid];
    else if (tid < 128) kt[tid - 64]  = k[off + tid - 64];
    else if (tid < 192) vt[tid - 128] = v[off + tid - 128];
    else                dt[tid - 192] = __expf(-__expf(w[off + tid - 192]));
    __syncthreads();

    float accv = 0.0f;
    const float vi = vt[i];
#pragma unroll
    for (int jj = 0; jj < 16; ++jj) {
      int j = jg * 16 + jj;
      float kv = kt[j] * vi;
      accv += rt[j] * (S[jj] + uj[jj] * kv);
      S[jj] = dt[j] * S[jj] + kv;
    }
    red[jg][i] = accv;
    __syncthreads();
    if (jg == 0) y[off + i] = red[0][i] + red[1][i] + red[2][i] + red[3][i];
    __syncthreads();
  }
}

// GroupNorm per head, * gamma + beta, * g(silu), emit bf16 for output GEMM
__global__ __launch_bounds__(64) void gn_mul_kernel(
    const float* __restrict__ y, const float* __restrict__ g,
    const float* __restrict__ gamma, const float* __restrict__ beta,
    __bf16* __restrict__ out) {
  const int tid = threadIdx.x;
  const size_t idx = (size_t)blockIdx.x * 64 + tid;
  float val = y[idx];
  __shared__ float s1[64], s2[64];
  s1[tid] = val; s2[tid] = val * val;
  __syncthreads();
  for (int s = 32; s > 0; s >>= 1) {
    if (tid < s) { s1[tid] += s1[tid + s]; s2[tid] += s2[tid + s]; }
    __syncthreads();
  }
  float mean = s1[0] * (1.0f / 64.0f);
  float var  = s2[0] * (1.0f / 64.0f) - mean * mean;
  int c = (blockIdx.x % RW_H) * 64 + tid;
  float yn = (val - mean) * rsqrtf(var + RW_EPS) * gamma[c] + beta[c];
  out[idx] = f2bf(yn * g[idx]);
}

// ---------------------------------------------------------------------------
// Host-side orchestration
// ---------------------------------------------------------------------------
static inline void launch_gemm(int epi, const __bf16* A, const __bf16* Bw,
                               float* C, const float* bias, int M, int N, int K,
                               hipStream_t s) {
  dim3 g((N + 255) / 256, (M + 127) / 128), b(256);
  bool exact = (M % 128 == 0) && (N % 256 == 0);
  if (exact) {
    if (epi == 1)      gemm_bf16_kernel<1, true><<<g, b, 0, s>>>(A, Bw, C, bias, M, N, K);
    else if (epi == 2) gemm_bf16_kernel<2, true><<<g, b, 0, s>>>(A, Bw, C, bias, M, N, K);
    else               gemm_bf16_kernel<0, true><<<g, b, 0, s>>>(A, Bw, C, bias, M, N, K);
  } else {
    if (epi == 1)      gemm_bf16_kernel<1, false><<<g, b, 0, s>>>(A, Bw, C, bias, M, N, K);
    else if (epi == 2) gemm_bf16_kernel<2, false><<<g, b, 0, s>>>(A, Bw, C, bias, M, N, K);
    else               gemm_bf16_kernel<0, false><<<g, b, 0, s>>>(A, Bw, C, bias, M, N, K);
  }
}

extern "C" void kernel_launch(void* const* d_in, const int* in_sizes, int n_in,
                              void* d_out, int out_size, void* d_ws, size_t ws_size,
                              hipStream_t stream) {
  const float* x      = (const float*)d_in[0];
  const float* maa_x  = (const float*)d_in[1];
  const float* maa_w  = (const float*)d_in[2];
  const float* maa_k  = (const float*)d_in[3];
  const float* maa_v  = (const float*)d_in[4];
  const float* maa_r  = (const float*)d_in[5];
  const float* maa_g  = (const float*)d_in[6];
  const float* w1mix  = (const float*)d_in[7];   // (C, 160)
  const float* w2mix  = (const float*)d_in[8];   // (5, 32, C) == (160, C)
  const float* tdecay = (const float*)d_in[9];   // (C)
  const float* dw1    = (const float*)d_in[10];  // (C, 64)
  const float* dw2    = (const float*)d_in[11];  // (64, C)
  const float* faaaa  = (const float*)d_in[12];  // (H, N)
  const float* W_r    = (const float*)d_in[13];  // (C, C)  [N,K]
  const float* W_k    = (const float*)d_in[14];
  const float* W_v    = (const float*)d_in[15];
  const float* W_g    = (const float*)d_in[16];
  const float* W_o    = (const float*)d_in[17];  // (C, C)  [N,K]
  const float* lng    = (const float*)d_in[18];
  const float* lnb    = (const float*)d_in[19];

  const size_t CC  = (size_t)RW_C * RW_C;        // 4.19M
  const size_t BTC = (size_t)RW_BT * RW_C;       // 16.8M

  char* p = (char*)d_ws;
  auto take = [&](size_t bytes) -> void* {
    void* r = (void*)p;
    p += (bytes + 255) & ~(size_t)255;
    return r;
  };

  __bf16* wr_bf   = (__bf16*)take(CC * 2);
  __bf16* wk_bf   = (__bf16*)take(CC * 2);
  __bf16* wv_bf   = (__bf16*)take(CC * 2);
  __bf16* wg_bf   = (__bf16*)take(CC * 2);
  __bf16* wo_bf   = (__bf16*)take(CC * 2);
  __bf16* w1T_bf  = (__bf16*)take((size_t)RW_DMIX5 * RW_C * 2);
  __bf16* dw1T_bf = (__bf16*)take((size_t)RW_DDEC * RW_C * 2);
  __bf16* dw2T_bf = (__bf16*)take((size_t)RW_C * RW_DDEC * 2);
  __bf16* xxx_bf  = (__bf16*)take(BTC * 2);
  float*  a_f     = (float*) take((size_t)RW_BT * RW_DMIX5 * 4);
  __bf16* xw_bf   = (__bf16*)take(BTC * 2);
  __bf16* xk_bf   = (__bf16*)take(BTC * 2);
  __bf16* xv_bf   = (__bf16*)take(BTC * 2);
  __bf16* xr_bf   = (__bf16*)take(BTC * 2);
  __bf16* xg_bf   = (__bf16*)take(BTC * 2);
  float*  r_f     = (float*) take(BTC * 4);
  float*  k_f     = (float*) take(BTC * 4);
  float*  v_f     = (float*) take(BTC * 4);
  float*  g_f     = (float*) take(BTC * 4);
  float*  ta_f    = (float*) take((size_t)RW_BT * RW_DDEC * 4);
  __bf16* ta_bf   = (__bf16*)take((size_t)RW_BT * RW_DDEC * 2);
  float*  w_f     = (float*) take(BTC * 4);
  float*  y_f     = (float*) take(BTC * 4);
  __bf16* yg_bf   = (__bf16*)take(BTC * 2);

  auto cvt = [&](const float* in, __bf16* out, long n) {
    cvt_f32_to_bf16_kernel<<<(unsigned)((n + 255) / 256), 256, 0, stream>>>(in, out, n);
  };
  auto cvtT = [&](const float* in, __bf16* out, int rows, int cols) {
    long n = (long)rows * cols;
    cvtT_f32_to_bf16_kernel<<<(unsigned)((n + 255) / 256), 256, 0, stream>>>(in, out, rows, cols);
  };

  // 1) weights -> bf16 (40 MB total; L2-resident at 192 MB)
  cvt(W_r, wr_bf, (long)CC);
  cvt(W_k, wk_bf, (long)CC);
  cvt(W_v, wv_bf, (long)CC);
  cvt(W_g, wg_bf, (long)CC);
  cvt(W_o, wo_bf, (long)CC);
  cvtT(w1mix, w1T_bf, RW_C, RW_DMIX5);   // -> [160, C]
  cvtT(dw1,   dw1T_bf, RW_C, RW_DDEC);   // -> [64, C]
  cvtT(dw2,   dw2T_bf, RW_DDEC, RW_C);   // -> [C, 64]

  // 2) token-shift lerp for the mix GEMM
  prep_xxx_kernel<<<(unsigned)((BTC + 255) / 256), 256, 0, stream>>>(
      x, maa_x, xxx_bf, (long)BTC);

  // 3) a = tanh(xxx @ w1)         [8192 x 160, K=2048]  (WMMA, tanh epilogue)
  launch_gemm(1, xxx_bf, w1T_bf, a_f, nullptr, RW_BT, RW_DMIX5, RW_C, stream);

  // 4) five mixed inputs in bf16
  mix5_kernel<<<RW_BT, 256, 0, stream>>>(x, a_f, w2mix, maa_w, maa_k, maa_v,
                                         maa_r, maa_g, xw_bf, xk_bf, xv_bf,
                                         xr_bf, xg_bf);

  // 5) the four big projections   [8192 x 2048, K=2048] (WMMA, EXACT path)
  launch_gemm(0, xr_bf, wr_bf, r_f, nullptr, RW_BT, RW_C, RW_C, stream);
  launch_gemm(0, xk_bf, wk_bf, k_f, nullptr, RW_BT, RW_C, RW_C, stream);
  launch_gemm(0, xv_bf, wv_bf, v_f, nullptr, RW_BT, RW_C, RW_C, stream);
  launch_gemm(2, xg_bf, wg_bf, g_f, nullptr, RW_BT, RW_C, RW_C, stream);  // silu

  // 6) decay path: tanh(xw @ dw1) @ dw2 + time_decay
  launch_gemm(1, xw_bf, dw1T_bf, ta_f, nullptr, RW_BT, RW_DDEC, RW_C, stream);
  cvt(ta_f, ta_bf, (long)RW_BT * RW_DDEC);
  launch_gemm(0, ta_bf, dw2T_bf, w_f, tdecay, RW_BT, RW_C, RW_DDEC, stream);

  // 7) sequential WKV6 scan: 128 (b,h) blocks, state in registers
  wkv6_scan_kernel<<<RW_B * RW_H, 256, 0, stream>>>(r_f, k_f, v_f, w_f, faaaa, y_f);

  // 8) groupnorm * g -> bf16
  gn_mul_kernel<<<(unsigned)((size_t)RW_BT * RW_H), 64, 0, stream>>>(
      y_f, g_f, lng, lnb, yg_bf);

  // 9) output projection           [8192 x 2048, K=2048] (WMMA, EXACT) -> d_out f32
  launch_gemm(0, yg_bf, wo_bf, (float*)d_out, nullptr, RW_BT, RW_C, RW_C, stream);
}
